// CausalSelfAttention_8761733284318
// MI455X (gfx1250) — compile-verified, matrix operand order checked by reading, not run on hardware
//
#include <hip/hip_runtime.h>
#include <hip/hip_bf16.h>

// ---------------------------------------------------------------------------
// CDNA5 (gfx1250, wave32) causal self-attention:
//   1) convert x, qkv_w, proj_w to bf16
//   2) qkv = x @ qkv_w^T + qkv_b          (WMMA bf16, f32 accum, bf16 out)
//   3) flash-attention per (b,h), causal  (WMMA bf16 QK^T and P·V, f32 softmax,
//      V tiles staged global->LDS by the Tensor Data Mover, double-buffered)
//   4) out = attn @ proj_w^T + proj_b     (WMMA bf16, f32 out)
// B=4, S=2048, C=1024, H=16, D=64
// ---------------------------------------------------------------------------

typedef __attribute__((ext_vector_type(16))) __bf16 bf16x16;
typedef __attribute__((ext_vector_type(8)))  __bf16 bf16x8;
typedef __attribute__((ext_vector_type(8)))  float  f32x8;
typedef __attribute__((ext_vector_type(4)))  unsigned int uint32x4;
typedef __attribute__((ext_vector_type(8)))  int int32x8;
typedef __attribute__((ext_vector_type(4)))  int int32x4;

union FragAB { bf16x16 v; bf16x8 h[2]; };
union FAcc   { f32x8 v; float f[8]; };

static __device__ __forceinline__ __bf16 f2bf(float f) {
    unsigned u = __builtin_bit_cast(unsigned, f);
    unsigned r = (u + 0x7FFFu + ((u >> 16) & 1u)) >> 16;
    unsigned short s = (unsigned short)r;
    return __builtin_bit_cast(__bf16, s);
}
static __device__ __forceinline__ unsigned short bfbits(__bf16 b) {
    return __builtin_bit_cast(unsigned short, b);
}

// Load one WMMA A/B fragment (16x32 / 32x16 bf16) from a k-contiguous row.
// Element i of lane L maps to k = k0 + (i>>3)*16 + (L>>4)*8 + (i&7):
// exactly two 16-byte chunks.
static __device__ __forceinline__ bf16x16 load_frag(const __bf16* rowptr, int k0, int hi) {
    FragAB f;
    f.h[0] = *(const bf16x8*)(rowptr + k0 + hi * 8);
    f.h[1] = *(const bf16x8*)(rowptr + k0 + 16 + hi * 8);
    return f.v;
}

static __device__ __forceinline__ f32x8 wmma_bf16(bf16x16 a, bf16x16 b, f32x8 c) {
    return __builtin_amdgcn_wmma_f32_16x16x32_bf16(false, a, false, b, (short)0, c,
                                                   false, false);
}

// XOR lane swap via ds_swizzle (group-of-32 mode: and=0x1f, or=0, xor=MASK).
template <int MASK>
static __device__ __forceinline__ float swzf(float x) {
    int i = __builtin_bit_cast(int, x);
    i = __builtin_amdgcn_ds_swizzle(i, (MASK << 10) | 0x1f);
    return __builtin_bit_cast(float, i);
}
static __device__ __forceinline__ float red_max16(float x) {
    x = fmaxf(x, swzf<8>(x)); x = fmaxf(x, swzf<4>(x));
    x = fmaxf(x, swzf<2>(x)); x = fmaxf(x, swzf<1>(x));
    return x;
}
static __device__ __forceinline__ float red_sum16(float x) {
    x += swzf<8>(x); x += swzf<4>(x); x += swzf<2>(x); x += swzf<1>(x);
    return x;
}

// ---------------------------------------------------------------------------
// Tensor Data Mover: DMA a 2-D tile (32 rows x 64 cols, 2-byte elements,
// row pitch 3072 elements) from global memory into LDS.  D# per
// CDNA5 ISA: group0 = {count, lds_addr, global_addr,
// type=2}; group1 = {data_size=2B, dims/tile/stride}; groups 2/3 zero (2-D).
// This toolchain's builtin takes 6 args (g0, g1, g2, g3, extra, cpol).
static __device__ __forceinline__ void tdm_load_v32x64(unsigned lds_off,
                                                       const __bf16* gptr) {
    const unsigned long long ga = (unsigned long long)(size_t)gptr;
    uint32x4 g0;
    g0[0] = 1u;                                        // count=1 (valid), user mode
    g0[1] = lds_off;                                   // lds_addr (bytes)
    g0[2] = (unsigned)ga;                              // global_addr[31:0]
    g0[3] = ((unsigned)(ga >> 32) & 0x1FFFFFFu) | (2u << 30);  // [56:32] | type=2
    int32x8 g1;
    g1[0] = 1 << 16;          // workgroup_mask=0, data_size=1 (2 bytes), no flags
    g1[1] = 64 << 16;         // tensor_dim0 = 64 (elements)
    g1[2] = 32 << 16;         // tensor_dim0[31:16]=0 | tensor_dim1 = 32
    g1[3] = 64 << 16;         // tensor_dim1[31:16]=0 | tile_dim0 = 64
    g1[4] = 32;               // tile_dim1 = 32, tile_dim2 = 0 (2-D)
    g1[5] = 3072;             // tensor_dim0_stride[31:0] = qkv row pitch (elems)
    g1[6] = 0;                // stride[47:32]=0 | tensor_dim1_stride lo = 0
    g1[7] = 0;
    const int32x4 z4 = {0, 0, 0, 0};
    const int32x8 z8 = {0, 0, 0, 0, 0, 0, 0, 0};
    __builtin_amdgcn_tensor_load_to_lds(g0, g1, z4, z4, z8, 0);
}

// ---------------------------------------------------------------------------
__global__ void cvt_bf16(const float* __restrict__ in, __bf16* __restrict__ out, int n) {
    int i = blockIdx.x * blockDim.x + threadIdx.x;
    int stride = gridDim.x * blockDim.x;
    for (; i < n; i += stride) out[i] = f2bf(in[i]);
}

// ---------------------------------------------------------------------------
// C[m][n] = sum_k A[m][k]*W[n][k] + bias[n].  WG = 8 waves, 128(M) x 256(N)
// tile; each wave owns a 64x64 tile = 4x4 WMMA accumulators. Operand
// fragments load straight from global (working set is L2-resident: 192MB L2).
// K-loop is software-pipelined: fragments for k+32 load while the 16 WMMAs
// for k execute; explicit prefetch pulls k+64 into the near cache.
template <bool OUT_BF16>
__global__ __launch_bounds__(256) void gemm_bias(
        const __bf16* __restrict__ A, const __bf16* __restrict__ W,
        const float* __restrict__ bias, void* __restrict__ outp,
        int M, int N, int K) {
    const int lane = threadIdx.x & 31, wave = threadIdx.x >> 5;
    const int r = lane & 15, hi = lane >> 4;
    const int m0 = blockIdx.y * 128 + (wave >> 2) * 64;
    const int n0 = blockIdx.x * 256 + (wave & 3) * 64;

    const __bf16* arow[4];
    const __bf16* wrow[4];
#pragma unroll
    for (int i = 0; i < 4; ++i) {
        arow[i] = A + (size_t)(m0 + i * 16 + r) * K;
        wrow[i] = W + (size_t)(n0 + i * 16 + r) * K;
    }

    FAcc acc[4][4] = {};
    bf16x16 af[2][4], bfr[2][4];
#pragma unroll
    for (int i = 0; i < 4; ++i) {
        af[0][i] = load_frag(arow[i], 0, hi);
        bfr[0][i] = load_frag(wrow[i], 0, hi);
    }

    for (int k0 = 0; k0 < K; k0 += 32) {
        const int cur = (k0 >> 5) & 1, nxt = cur ^ 1;
        if (k0 + 32 < K) {
#pragma unroll
            for (int i = 0; i < 4; ++i) {
                af[nxt][i] = load_frag(arow[i], k0 + 32, hi);
                bfr[nxt][i] = load_frag(wrow[i], k0 + 32, hi);
            }
            // gfx1250 global_prefetch_b8: warm the near cache 2 steps ahead
            if (k0 + 64 < K) {
#pragma unroll
                for (int i = 0; i < 4; ++i) {
                    __builtin_prefetch(arow[i] + k0 + 64 + hi * 8, 0, 0);
                    __builtin_prefetch(wrow[i] + k0 + 64 + hi * 8, 0, 0);
                }
            }
        }
#pragma unroll
        for (int i = 0; i < 4; ++i)
#pragma unroll
            for (int j = 0; j < 4; ++j)
                acc[i][j].v = wmma_bf16(af[cur][i], bfr[cur][j], acc[i][j].v);
    }

#pragma unroll
    for (int j = 0; j < 4; ++j) {
        const int n = n0 + j * 16 + r;
        const float bb = bias[n];
#pragma unroll
        for (int i = 0; i < 4; ++i) {
#pragma unroll
            for (int v = 0; v < 8; ++v) {
                const float val = acc[i][j].f[v] + bb;
                const int row = m0 + i * 16 + v + 8 * hi;   // C-layout: M = v + 8*hi
                if (OUT_BF16)
                    ((__bf16*)outp)[(size_t)row * N + n] = f2bf(val);
                else
                    ((float*)outp)[(size_t)row * N + n] = val;
            }
        }
    }
}

// ---------------------------------------------------------------------------
// Flash attention. qkv: [B*S][3072] bf16 (Q at +0, K at +1024, V at +2048,
// head h at +h*64). Grid: x = query block of 64 (32), y = b*16+h (64).
// 4 waves of 32; wave w owns query rows r0..r0+15; key blocks of 32.
// V tiles are DMA'd global->LDS by the TDM, double-buffered so the copy for
// block kb+1 runs under the softmax/WMMA work of block kb.
#define SD 40   // LDS row stride in bf16 elements (80B: 16B-aligned, conflict-free)

__global__ __launch_bounds__(128) void attn_kernel(const __bf16* __restrict__ qkv,
                                                   __bf16* __restrict__ out) {
    __shared__ __align__(16) unsigned short ldsVraw[2 * 32 * 64]; // TDM dst [key][d]
    __shared__ __align__(16) unsigned short ldsVT[64 * SD];       // V^T : [d][key]
    __shared__ __align__(16) unsigned short ldsP[4 * 16 * SD];    // per-wave P tile

    const int tid = threadIdx.x;
    const int wave = tid >> 5, lane = tid & 31;
    const int r = lane & 15, hi = lane >> 4;
    const int wgm = blockIdx.x;
    const int b = blockIdx.y >> 4, h = blockIdx.y & 15;
    const int r0 = wgm * 64 + wave * 16;
    const size_t RS = 3072;                 // qkv row stride (elements)
    const size_t bbase = (size_t)b * 2048;

    // generic->LDS offset: low 32 bits of the shared aperture address
    const unsigned vraw_off = (unsigned)(size_t)(const void*)&ldsVraw[0];
    const __bf16* vbase = qkv + bbase * RS + 2048 + h * 64;   // V, this (b,h)

    // Q A-fragments (reused across all key blocks): k-dim = head dim 0..63
    const __bf16* qrow = qkv + (bbase + r0 + r) * RS + h * 64;
    const bf16x16 qa0 = load_frag(qrow, 0, hi);
    const bf16x16 qa1 = load_frag(qrow, 32, hi);

    FAcc o[4] = {};
    float mi[8], li[8];
#pragma unroll
    for (int v = 0; v < 8; ++v) { mi[v] = -1e30f; li[v] = 0.0f; }

    const int kbmax = 2 * wgm + 2;

    // prologue: DMA the first V tile
    if (wave == 0) tdm_load_v32x64(vraw_off, vbase);

    for (int kb = 0; kb < kbmax; ++kb) {
        const int cur = kb & 1;
        // issue DMA for the next tile, then wait for the current one
        if (wave == 0) {
            if (kb + 1 < kbmax) {
                tdm_load_v32x64(vraw_off + (unsigned)((cur ^ 1) * 32 * 64 * 2),
                                vbase + (size_t)(kb + 1) * 32 * RS);
                __builtin_amdgcn_s_wait_tensorcnt(1);
            } else {
                __builtin_amdgcn_s_wait_tensorcnt(0);
            }
        }
        __syncthreads();    // A: ldsVraw[cur] visible to all waves

        // ---- transpose raw V tile [key][d] -> V^T [d][key] in LDS ----
        {
            const int keyl = tid & 31;
            const int dbase = (tid >> 5) * 16;
            const unsigned short* vr = ldsVraw + cur * (32 * 64) + keyl * 64 + dbase;
            const bf16x8 c0 = *(const bf16x8*)vr;
            const bf16x8 c1 = *(const bf16x8*)(vr + 8);
#pragma unroll
            for (int j = 0; j < 8; ++j) {
                ldsVT[(dbase + j) * SD + keyl] = bfbits(c0[j]);
                ldsVT[(dbase + 8 + j) * SD + keyl] = bfbits(c1[j]);
            }
        }
        __syncthreads();    // B: V^T ready

        if (kb * 32 <= r0 + 15) {   // wave-uniform: EXEC stays all-ones inside
            // ---- scores: load all K fragments first, then 4 WMMAs ----
            const __bf16* krow0 = qkv + (bbase + kb * 32 + r) * RS + 1024 + h * 64;
            const __bf16* krow1 = krow0 + 16 * RS;
            const bf16x16 k00 = load_frag(krow0, 0, hi);
            const bf16x16 k01 = load_frag(krow0, 32, hi);
            const bf16x16 k10 = load_frag(krow1, 0, hi);
            const bf16x16 k11 = load_frag(krow1, 32, hi);
            FAcc s0 = {}, s1 = {};
            s0.v = wmma_bf16(qa0, k00, s0.v);
            s0.v = wmma_bf16(qa1, k01, s0.v);
            s1.v = wmma_bf16(qa0, k10, s1.v);
            s1.v = wmma_bf16(qa1, k11, s1.v);

            // ---- causal mask + online softmax (C-layout: row = r0+v+8*hi) ----
            unsigned short* P = ldsP + wave * 16 * SD;
            const int c0i = kb * 32 + r, c1i = kb * 32 + 16 + r;
#pragma unroll
            for (int v = 0; v < 8; ++v) {
                const int row = r0 + v + 8 * hi;
                float a0 = s0.f[v] * 0.125f;          // 1/sqrt(64)
                float a1 = s1.f[v] * 0.125f;
                if (c0i > row) a0 = -1e30f;
                if (c1i > row) a1 = -1e30f;
                const float mtile = red_max16(fmaxf(a0, a1));
                const float mnew = fmaxf(mi[v], mtile);
                const float alpha = __expf(mi[v] - mnew);
                const float e0 = __expf(a0 - mnew);
                const float e1 = __expf(a1 - mnew);
                li[v] = li[v] * alpha + red_sum16(e0 + e1);
                mi[v] = mnew;
#pragma unroll
                for (int t = 0; t < 4; ++t) o[t].f[v] *= alpha;
                P[(v + 8 * hi) * SD + r] = bfbits(f2bf(e0));
                P[(v + 8 * hi) * SD + 16 + r] = bfbits(f2bf(e1));
            }
            // ---- O += P (16x32, A-layout via LDS) x V (32x64, B from V^T) ----
            const bf16x16 pa = load_frag((const __bf16*)P + r * SD, 0, hi);
            const bf16x16 vb0 = load_frag((const __bf16*)ldsVT + (0 * 16 + r) * SD, 0, hi);
            const bf16x16 vb1 = load_frag((const __bf16*)ldsVT + (1 * 16 + r) * SD, 0, hi);
            const bf16x16 vb2 = load_frag((const __bf16*)ldsVT + (2 * 16 + r) * SD, 0, hi);
            const bf16x16 vb3 = load_frag((const __bf16*)ldsVT + (3 * 16 + r) * SD, 0, hi);
            o[0].v = wmma_bf16(pa, vb0, o[0].v);
            o[1].v = wmma_bf16(pa, vb1, o[1].v);
            o[2].v = wmma_bf16(pa, vb2, o[2].v);
            o[3].v = wmma_bf16(pa, vb3, o[3].v);
        }
        __syncthreads();    // C: V^T free for next iteration's transpose
    }

    // ---- normalize and write attn output (bf16) ----
#pragma unroll
    for (int v = 0; v < 8; ++v) {
        const float inv = 1.0f / li[v];
        const int row = r0 + v + 8 * hi;
        const size_t base = (bbase + row) * 1024 + h * 64;
#pragma unroll
        for (int t = 0; t < 4; ++t)
            out[base + t * 16 + r] = f2bf(o[t].f[v] * inv);
    }
}

// ---------------------------------------------------------------------------
extern "C" void kernel_launch(void* const* d_in, const int* in_sizes, int n_in,
                              void* d_out, int out_size, void* d_ws, size_t ws_size,
                              hipStream_t stream) {
    const float* x      = (const float*)d_in[0];
    const float* qkv_w  = (const float*)d_in[1];
    const float* qkv_b  = (const float*)d_in[2];
    const float* proj_w = (const float*)d_in[3];
    const float* proj_b = (const float*)d_in[4];

    char* ws = (char*)d_ws;
    __bf16* x_bf   = (__bf16*)ws; ws += (size_t)8192 * 1024 * 2;
    __bf16* qw_bf  = (__bf16*)ws; ws += (size_t)3072 * 1024 * 2;
    __bf16* pw_bf  = (__bf16*)ws; ws += (size_t)1024 * 1024 * 2;
    __bf16* qkv_bf = (__bf16*)ws; ws += (size_t)8192 * 3072 * 2;
    __bf16* at_bf  = (__bf16*)ws;

    cvt_bf16<<<2048, 256, 0, stream>>>(x, x_bf, 8192 * 1024);
    cvt_bf16<<<1024, 256, 0, stream>>>(qkv_w, qw_bf, 3072 * 1024);
    cvt_bf16<<<512, 256, 0, stream>>>(proj_w, pw_bf, 1024 * 1024);

    // qkv = x @ qkv_w^T + qkv_b  -> bf16 [8192][3072]
    gemm_bias<true><<<dim3(12, 64), 256, 0, stream>>>(x_bf, qw_bf, qkv_b,
                                                      (void*)qkv_bf, 8192, 3072, 1024);
    // causal flash attention -> bf16 [8192][1024]
    attn_kernel<<<dim3(32, 64), 128, 0, stream>>>(qkv_bf, at_bf);
    // out = attn @ proj_w^T + proj_b -> f32 d_out
    gemm_bias<false><<<dim3(4, 64), 256, 0, stream>>>(at_bf, pw_bf, proj_b,
                                                      d_out, 8192, 1024, 1024);
}
